// VQLayer_77438260347297
// MI455X (gfx1250) — matrix operand order repaired and checked by reading, not compiled, not run on hardware
//
#include <hip/hip_runtime.h>
#include <stdint.h>

typedef __attribute__((ext_vector_type(16))) __bf16 v16bf;
typedef __attribute__((ext_vector_type(8)))  float  v8f;

#define B_ROWS 16384
#define K_PROT 1024
#define D_DIM  512
#define MTILE  32           // rows per workgroup
#define APITCH 520          // bank-conflict-free LDS pitch (bf16 elems)
#define EPSF   1e-6f
#define ENTW   0.1f

__device__ __forceinline__ uint16_t f32_to_bf16(float f) {
    uint32_t x = __float_as_uint(f);
    uint32_t r = x + 0x7fffu + ((x >> 16) & 1u);   // round-to-nearest-even
    return (uint16_t)(r >> 16);
}

__device__ __forceinline__ float hash_u01(uint32_t x) {
    x ^= x >> 16; x *= 0x7feb352du;
    x ^= x >> 15; x *= 0x846ca68bu;
    x ^= x >> 16;
    return (float)(x >> 8) * (1.0f / 16777216.0f) + 1e-12f;
}

// ---------- prep: X -> bf16 + row norms (1 wave per row) ----------
__global__ void prep_latents(const float* __restrict__ X,
                             uint16_t* __restrict__ Xb,
                             float* __restrict__ xnorm) {
    int wave = (blockIdx.x * blockDim.x + threadIdx.x) >> 5;
    int lane = threadIdx.x & 31;
    if (wave >= B_ROWS) return;
    const float* row = X + (size_t)wave * D_DIM;
    uint16_t*   orow = Xb + (size_t)wave * D_DIM;
    int c0 = lane * 16;
    float s = 0.f;
#pragma unroll
    for (int i = 0; i < 16; ++i) {
        float v = row[c0 + i];
        s += v * v;
        orow[c0 + i] = f32_to_bf16(v);
    }
#pragma unroll
    for (int m = 16; m >= 1; m >>= 1) s += __shfl_xor(s, m, 32);
    if (lane == 0) xnorm[wave] = s;
}

// ---------- prep: P -> bf16 row-major (PT[k][d]) + norms ----------
__global__ void prep_protos(const float* __restrict__ P,
                            uint16_t* __restrict__ PT,
                            float* __restrict__ pnorm) {
    int wave = (blockIdx.x * blockDim.x + threadIdx.x) >> 5;
    int lane = threadIdx.x & 31;
    if (wave >= K_PROT) return;
    const float* row = P + (size_t)wave * D_DIM;
    int c0 = lane * 16;
    float s = 0.f;
#pragma unroll
    for (int i = 0; i < 16; ++i) {
        float v = row[c0 + i];
        s += v * v;
        PT[(size_t)wave * D_DIM + c0 + i] = f32_to_bf16(v);
    }
#pragma unroll
    for (int m = 16; m >= 1; m >>= 1) s += __shfl_xor(s, m, 32);
    if (lane == 0) pnorm[wave] = s;
}

// ---------- zero the atomic accumulators (graph-replay safe) ----------
__global__ void zero_accum(float* __restrict__ colsum, float* __restrict__ sacc) {
    for (int i = threadIdx.x; i < K_PROT; i += blockDim.x) colsum[i] = 0.f;
    if (threadIdx.x == 0) sacc[0] = 0.f;
}

// ---------- main fused kernel: 32 rows per WG, 8 waves x 128 cols ----------
__global__ __launch_bounds__(256) void vq_main(
        const uint16_t* __restrict__ Xb, const uint16_t* __restrict__ PT,
        const float* __restrict__ xnorm, const float* __restrict__ pnorm,
        const float* __restrict__ Pf32,
        float* __restrict__ out, float* __restrict__ colsum,
        float* __restrict__ sacc)
{
    __shared__ uint16_t As[MTILE * APITCH];
    __shared__ float redF[8][MTILE];
    __shared__ int   redI[8][MTILE];
    __shared__ float rowStat[MTILE];   // rowmax, then logZ
    __shared__ int   rowIdx[MTILE];
    __shared__ float slpRed[8];

    const int tid  = threadIdx.x;
    const int wave = tid >> 5;
    const int lane = tid & 31;
    const int hl   = lane >> 4;      // half-lane group (0/1)
    const int ll   = lane & 15;
    const int row0 = blockIdx.x * MTILE;
    const int colBase = wave * 128;

    // stage A block 32 x 512 bf16 -> LDS (64 bf16 = 128B per thread)
    {
        int base = tid * 64;
        int r = base >> 9, c = base & 511;
        const uint4* g = (const uint4*)(Xb + (size_t)(row0 + r) * D_DIM + c);
        uint4* s = (uint4*)(&As[r * APITCH + c]);
#pragma unroll
        for (int q = 0; q < 8; ++q) s[q] = g[q];
    }
    __syncthreads();

    v8f acc[2][8];
#pragma unroll
    for (int g = 0; g < 2; ++g)
#pragma unroll
        for (int t = 0; t < 8; ++t)
            acc[g][t] = (v8f){0.f,0.f,0.f,0.f,0.f,0.f,0.f,0.f};

    for (int kk = 0; kk < D_DIM; kk += 32) {
        // two A fragments (row groups 0-15, 16-31) from LDS
        union { uint4 u[2]; v16bf v; } af0, af1;
        const int koff = kk + (hl << 3);
        const uint16_t* ap0 = &As[ll * APITCH + koff];
        const uint16_t* ap1 = &As[(16 + ll) * APITCH + koff];
        af0.u[0] = *(const uint4*)ap0;
        af0.u[1] = *(const uint4*)(ap0 + 16);
        af1.u[0] = *(const uint4*)ap1;
        af1.u[1] = *(const uint4*)(ap1 + 16);

        const uint16_t* bp = PT + (size_t)(colBase + ll) * D_DIM + kk + (hl << 3);
        if (kk + 32 < D_DIM)
            __builtin_prefetch(bp + 32, 0, 1);   // global_prefetch_b8

#pragma unroll
        for (int t = 0; t < 8; ++t) {
            union { uint4 u[2]; v16bf v; } bf;
            bf.u[0] = *(const uint4*)(bp + t * 16 * D_DIM);
            bf.u[1] = *(const uint4*)(bp + t * 16 * D_DIM + 16);
            // each B fragment feeds two WMMAs (row groups) -> 2x intensity
            acc[0][t] = __builtin_amdgcn_wmma_f32_16x16x32_bf16(
                false, af0.v, false, bf.v, (short)0, acc[0][t], false, false);
            acc[1][t] = __builtin_amdgcn_wmma_f32_16x16x32_bf16(
                false, af1.v, false, bf.v, (short)0, acc[1][t], false, false);
        }
    }

    // logits = 2*(X.P^T) - ||x||^2 - ||p||^2, in place
    {
        float pnr[8];
#pragma unroll
        for (int t = 0; t < 8; ++t) pnr[t] = pnorm[colBase + t * 16 + ll];
#pragma unroll
        for (int g = 0; g < 2; ++g)
#pragma unroll
            for (int j = 0; j < 8; ++j) {
                float xn = xnorm[row0 + g * 16 + hl * 8 + j];
#pragma unroll
                for (int t = 0; t < 8; ++t)
                    acc[g][t][j] = 2.0f * acc[g][t][j] - xn - pnr[t];
            }
    }

    // ---- row max (over K=1024, split across 8 waves) ----
#pragma unroll
    for (int g = 0; g < 2; ++g)
#pragma unroll
        for (int j = 0; j < 8; ++j) {
            float m = -3.4e38f;
#pragma unroll
            for (int t = 0; t < 8; ++t) m = fmaxf(m, acc[g][t][j]);
#pragma unroll
            for (int s = 1; s <= 8; s <<= 1) m = fmaxf(m, __shfl_xor(m, s, 32));
            if (ll == 0) redF[wave][g * 16 + hl * 8 + j] = m;
        }
    __syncthreads();
    if (tid < MTILE) {
        float m = -3.4e38f;
#pragma unroll
        for (int w = 0; w < 8; ++w) m = fmaxf(m, redF[w][tid]);
        rowStat[tid] = m;
    }
    __syncthreads();

    // ---- row sum-exp ----
#pragma unroll
    for (int g = 0; g < 2; ++g)
#pragma unroll
        for (int j = 0; j < 8; ++j) {
            float m = rowStat[g * 16 + hl * 8 + j];
            float s = 0.f;
#pragma unroll
            for (int t = 0; t < 8; ++t) s += __expf(acc[g][t][j] - m);
#pragma unroll
            for (int m2 = 1; m2 <= 8; m2 <<= 1) s += __shfl_xor(s, m2, 32);
            if (ll == 0) redF[wave][g * 16 + hl * 8 + j] = s;
        }
    __syncthreads();
    if (tid < MTILE) {
        float s = 0.f;
#pragma unroll
        for (int w = 0; w < 8; ++w) s += redF[w][tid];
        rowStat[tid] = rowStat[tid] + __logf(s);   // logZ
    }
    __syncthreads();

    // ---- logprob stats, column prob sums, gumbel argmax (streamed per row) ----
    float slp = 0.f;
    float cs[8];
#pragma unroll
    for (int t = 0; t < 8; ++t) cs[t] = 0.f;

#pragma unroll
    for (int g = 0; g < 2; ++g)
#pragma unroll
        for (int j = 0; j < 8; ++j) {
            const int rloc = g * 16 + hl * 8 + j;
            const int row  = row0 + rloc;
            const float lz = rowStat[rloc];
            float bv = -3.4e38f; int bc = 0;
#pragma unroll
            for (int t = 0; t < 8; ++t) {
                int col = colBase + t * 16 + ll;
                float lp = acc[g][t][j] - lz;
                slp += lp;
                cs[t] += __expf(lp);
                uint32_t h = (uint32_t)(row * K_PROT + col) * 2654435761u + 0x42424242u;
                float gn = -__logf(-__logf(hash_u01(h)));
                float y = acc[g][t][j] + gn;            // tau = 1
                if (y > bv || (y == bv && col < bc)) { bv = y; bc = col; }
            }
            // in-wave argmax reduce for this row (keeps register pressure low)
#pragma unroll
            for (int s = 1; s <= 8; s <<= 1) {
                float ov = __shfl_xor(bv, s, 32);
                int   oc = __shfl_xor(bc, s, 32);
                if (ov > bv || (ov == bv && oc < bc)) { bv = ov; bc = oc; }
            }
            if (ll == 0) { redF[wave][rloc] = bv; redI[wave][rloc] = bc; }
        }
    __syncthreads();
    if (tid < MTILE) {
        float bv = -3.4e38f; int bc = 0;
#pragma unroll
        for (int w = 0; w < 8; ++w) {
            float v = redF[w][tid]; int c = redI[w][tid];
            if (v > bv || (v == bv && c < bc)) { bv = v; bc = c; }
        }
        rowIdx[tid] = bc;
    }

    // sum-of-logprobs -> wave partials
#pragma unroll
    for (int s = 1; s <= 16; s <<= 1) slp += __shfl_xor(slp, s, 32);
    if (lane == 0) slpRed[wave] = slp;
    __syncthreads();
    if (tid == 0) {
        float tot = 0.f;
#pragma unroll
        for (int w = 0; w < 8; ++w) tot += slpRed[w];
        atomicAdd(sacc, tot);
    }

    // column prob sums -> true_prior accumulation
#pragma unroll
    for (int t = 0; t < 8; ++t) {
        float v = cs[t] + __shfl_xor(cs[t], 16, 32);
        if (lane < 16) atomicAdd(&colsum[colBase + t * 16 + lane], v);
    }

    // gather quantized rows: out[row0+r] = P[rowIdx[r]]
    for (int i = tid; i < MTILE * 128; i += 256) {
        int r = i >> 7, c = i & 127;
        int src = rowIdx[r];
        float4 v = ((const float4*)(Pf32 + (size_t)src * D_DIM))[c];
        ((float4*)(out + (size_t)(row0 + r) * D_DIM))[c] = v;
    }
}

// ---------- finalize: entropy + capacity + total ----------
__global__ void vq_finalize(const float* __restrict__ colsum,
                            const float* __restrict__ sacc,
                            float* __restrict__ out2) {
    __shared__ float red[8];
    int tid = threadIdx.x, lane = tid & 31, wave = tid >> 5;
    float e = 0.f;
    for (int k = tid; k < K_PROT; k += 256) {
        float tp = colsum[k] * (1.0f / B_ROWS) + EPSF;
        e -= tp * __logf(tp);
    }
#pragma unroll
    for (int s = 1; s <= 16; s <<= 1) e += __shfl_xor(e, s, 32);
    if (lane == 0) red[wave] = e;
    __syncthreads();
    if (tid == 0) {
        float ent = 0.f;
#pragma unroll
        for (int w = 0; w < 8; ++w) ent += red[w];
        float S = sacc[0];
        float capacity = __logf(1.0f / K_PROT) - S / ((float)B_ROWS * (float)K_PROT);
        float total = capacity + ENTW * ent;     // KL_WEIGHT = 1
        out2[0] = total;
        out2[1] = capacity;
    }
}

extern "C" void kernel_launch(void* const* d_in, const int* in_sizes, int n_in,
                              void* d_out, int out_size, void* d_ws, size_t ws_size,
                              hipStream_t stream) {
    const float* X = (const float*)d_in[0];   // [16384, 512]
    const float* P = (const float*)d_in[1];   // [1024, 512]
    float* out = (float*)d_out;               // [16384*512] + total + capacity

    char* ws = (char*)d_ws;
    uint16_t* Xb    = (uint16_t*)(ws);                               // 16 MiB
    uint16_t* PT    = (uint16_t*)(ws + (size_t)B_ROWS * D_DIM * 2);  // 1 MiB
    float*    xnorm = (float*)(ws + 17825792);                       // 64 KiB
    float*    pnorm = (float*)(ws + 17891328);                       // 4 KiB
    float*    csum  = (float*)(ws + 17895424);                       // 4 KiB
    float*    sacc  = (float*)(ws + 17899520);                       // 4 B

    zero_accum<<<1, 256, 0, stream>>>(csum, sacc);
    prep_latents<<<B_ROWS / 8, 256, 0, stream>>>(X, Xb, xnorm);
    prep_protos<<<K_PROT / 8, 256, 0, stream>>>(P, PT, pnorm);
    vq_main<<<B_ROWS / MTILE, 256, 0, stream>>>(Xb, PT, xnorm, pnorm, P,
                                                out, csum, sacc);
    vq_finalize<<<1, 256, 0, stream>>>(csum, sacc, out + (size_t)B_ROWS * D_DIM);
}
